// MultiScaleGNN_84653805404493
// MI455X (gfx1250) — matrix operand order repaired and checked by reading, not compiled, not run on hardware
//
#include <hip/hip_runtime.h>
#include <hip/hip_bf16.h>

// ---------------------------------------------------------------------------
// Types for CDNA5 WMMA (wave32): v16bf A/B fragments, v8f f32 accumulator.
// ---------------------------------------------------------------------------
typedef __attribute__((ext_vector_type(16))) __bf16 v16bf;
typedef __attribute__((ext_vector_type(8)))  float  v8f;

#define EPS 1e-5f

// pack two fp32 -> one dword of two bf16 (round-to-nearest-even), lo|hi<<16
__device__ __forceinline__ unsigned int pk2bf(float a, float b) {
  unsigned int ua = __builtin_bit_cast(unsigned int, a);
  unsigned int ub = __builtin_bit_cast(unsigned int, b);
  ua = (ua + 0x7FFFu + ((ua >> 16) & 1u)) >> 16;
  ub = (ub + 0x7FFFu + ((ub >> 16) & 1u)) >> 16;
  return ua | (ub << 16);
}

union FragU { uint4 q[2]; v16bf v; };

// ---------------------------------------------------------------------------
// GEMM: H[M,Nc] = X[M,K] @ W[K,Nc]   (fp32 in, bf16 WMMA, fp32 accumulate/out)
//
// Block: 256 threads = 8 waves. Block tile 128(M) x 64(N). K-step 32.
// Wave w owns M-subtile w (16 rows) and iterates 4 N-subtiles -> 4 WMMAs/step.
//
// Out-of-range M rows are CLAMPED to row M-1 during staging (branch-free):
// tile row r only ever feeds output row m0+r, which is never stored when
// m0+r >= M, so duplicated data in padding rows is harmless.
//
// LDS holds bf16 pairs packed in dwords:
//   As32[128][16] : As32[m][k/2]  (row-major; A frag = 2 contiguous uint4)
//   Bs32[64][16]  : Bs32[n][k/2]  (transposed; B frag = 2 contiguous uint4)
//
// Fragment layouts per CDNA5 ISA 7.12.2 (16-bit, 16x16x32):
//   A: lanes 0-15 row=lane, e0..7 -> K=0..7,  e8..15 -> K=16..23
//      lanes16-31 row=lane-16, e0..7 -> K=8..15, e8..15 -> K=24..31
//      => dwords [half*4 .. +3] and [8+half*4 .. +3] of row
//   B: lanes 0-15 col=lane, lanes16-31 col=lane-16, same K order as A
//      => dwords [half*4 .. +3] and [8+half*4 .. +3] of transposed row
//   C/D: VGPR r -> M = half*8 + r, N = lane&15
// ---------------------------------------------------------------------------
#define MT 128
#define NT 64
#define KT 32

__global__ __launch_bounds__(256) void gcn_gemm_wmma_bf16(
    const float* __restrict__ X, const float* __restrict__ W,
    float* __restrict__ H, int M, int K, int Nc)
{
  __shared__ unsigned int As32[MT][KT / 2];   // 8 KB
  __shared__ unsigned int Bs32[NT][KT / 2];   // 4 KB

  const int m0   = blockIdx.x * MT;
  const int n0   = blockIdx.y * NT;
  const int tid  = threadIdx.x;
  const int wave = tid >> 5;                  // 0..7 -> M subtile
  const int lane = tid & 31;
  const int half = lane >> 4;                 // K-half selector
  const int l16  = lane & 15;

  // branch-free clamped row addresses for this thread's staging slots
  const int ar = tid >> 4;                    // A rows: tid/16, +16 per iter
  const int ac = tid & 15;
  const int bn = tid & 63;                    // B col
  const int bk = tid >> 6;                    // B K-pair, +4 per iter

  v8f acc[4] = {{}, {}, {}, {}};

  for (int kk = 0; kk < K; kk += KT) {
    // ---- stage A tile: 128 rows x 16 dwords, 8 per thread (branch-free) ----
    #pragma unroll
    for (int it = 0; it < (MT * KT / 2) / 256; ++it) {
      int r  = ar + it * 16;
      int gm = min(m0 + r, M - 1);            // clamp instead of predicate
      float2 f = *(const float2*)&X[(size_t)gm * K + kk + 2 * ac];
      As32[r][ac] = pk2bf(f.x, f.y);
    }
    // ---- stage B tile transposed: 64 cols x 16 dwords, 4 per thread --------
    // consecutive threads -> consecutive n => both global reads coalesced
    #pragma unroll
    for (int it = 0; it < (NT * KT / 2) / 256; ++it) {
      int kp = bk + it * 4;
      const float* wp = &W[(size_t)(kk + 2 * kp) * Nc + n0 + bn];
      Bs32[bn][kp] = pk2bf(wp[0], wp[Nc]);
    }
    // prefetch next K-slab of X while we compute on this one
    if (kk + KT < K) {
      int gm = min(m0 + (tid >> 1), M - 1);
      __builtin_prefetch(&X[(size_t)gm * K + kk + KT], 0, 1);
    }
    __syncthreads();

    // ---- A fragment: two ds_load_b128 from this wave's 16-row subtile ------
    FragU ua;
    const unsigned int* arow = &As32[wave * 16 + l16][0];
    ua.q[0] = *(const uint4*)&arow[half * 4];
    ua.q[1] = *(const uint4*)&arow[8 + half * 4];
    const v16bf a = ua.v;

    // ---- 4 N-subtiles: B fragment (2x ds_load_b128) + WMMA -----------------
    #pragma unroll
    for (int nt = 0; nt < 4; ++nt) {
      FragU ub;
      const unsigned int* brow = &Bs32[nt * 16 + l16][0];
      ub.q[0] = *(const uint4*)&brow[half * 4];
      ub.q[1] = *(const uint4*)&brow[8 + half * 4];
      acc[nt] = __builtin_amdgcn_wmma_f32_16x16x32_bf16(
          false, a, false, ub.v, (short)0, acc[nt], false, false);
    }
    __syncthreads();
  }

  // ---- store D fragments ---------------------------------------------------
  #pragma unroll
  for (int r = 0; r < 8; ++r) {
    int gm = m0 + wave * 16 + half * 8 + r;
    if (gm < M) {
      size_t base = (size_t)gm * Nc + n0 + l16;
      #pragma unroll
      for (int nt = 0; nt < 4; ++nt) H[base + nt * 16] = acc[nt][r];
    }
  }
}

// ---------------------------------------------------------------------------
// Degree / dinv kernels (deg includes +1 self-loop; dinv = rsqrt(deg) in place)
// ---------------------------------------------------------------------------
__global__ void deg_init(float* deg, int N) {
  int i = blockIdx.x * blockDim.x + threadIdx.x;
  if (i < N) deg[i] = 1.0f;
}
__global__ void deg_accum(const int* __restrict__ dst, int E, float* deg) {
  int e = blockIdx.x * blockDim.x + threadIdx.x;
  if (e < E) atomicAdd(&deg[dst[e]], 1.0f);
}
__global__ void deg_rsqrt(float* deg, int N) {
  int i = blockIdx.x * blockDim.x + threadIdx.x;
  if (i < N) deg[i] = rsqrtf(deg[i]);
}

// ---------------------------------------------------------------------------
// agg[i,f] = h[i,f] * dinv[i]^2   (self-loop term; also zero-initializes agg)
// ---------------------------------------------------------------------------
__global__ void agg_selfloop(const float* __restrict__ H,
                             const float* __restrict__ dinv,
                             float* __restrict__ AGG, int N, int lgD) {
  long long idx = (long long)blockIdx.x * blockDim.x + threadIdx.x;
  if (idx < ((long long)N << lgD)) {
    int i = (int)(idx >> lgD);
    float d = dinv[i];
    AGG[idx] = H[idx] * d * d;
  }
}

// ---------------------------------------------------------------------------
// edge scatter: AGG[dst,f] += H[src,f] * dinv[src]*dinv[dst]
// ---------------------------------------------------------------------------
__global__ void agg_scatter(const float* __restrict__ H,
                            const float* __restrict__ dinv,
                            const int* __restrict__ src,
                            const int* __restrict__ dst,
                            float* __restrict__ AGG, long long E, int lgD) {
  long long idx = (long long)blockIdx.x * blockDim.x + threadIdx.x;
  long long total = E << lgD;
  if (idx < total) {
    long long e = idx >> lgD;
    int f = (int)(idx & ((1 << lgD) - 1));
    int s = src[e], d = dst[e];
    float nrm = dinv[s] * dinv[d];
    atomicAdd(&AGG[((long long)d << lgD) + f], H[((long long)s << lgD) + f] * nrm);
  }
}

// ---------------------------------------------------------------------------
// Y = act((AGG + bias - m) * rsqrt(v+eps) * g + bt)
// ---------------------------------------------------------------------------
__global__ void bias_bn_act(const float* __restrict__ AGG,
                            const float* __restrict__ Bv,
                            const float* __restrict__ g,
                            const float* __restrict__ bt,
                            const float* __restrict__ m,
                            const float* __restrict__ v,
                            float* __restrict__ Y, int N, int lgD, int relu) {
  long long idx = (long long)blockIdx.x * blockDim.x + threadIdx.x;
  if (idx < ((long long)N << lgD)) {
    int f = (int)(idx & ((1 << lgD) - 1));
    float x = AGG[idx] + Bv[f];
    float y = (x - m[f]) * rsqrtf(v[f] + EPS) * g[f] + bt[f];
    Y[idx] = relu ? fmaxf(y, 0.0f) : y;
  }
}

// ---------------------------------------------------------------------------
// column sum of emb [N,64] into out[64] (out must be pre-zeroed)
// ---------------------------------------------------------------------------
__global__ __launch_bounds__(256) void colsum64(const float* __restrict__ emb,
                                                float* __restrict__ out, int N) {
  __shared__ float s[256];
  float acc = 0.0f;
  long long total = (long long)N * 64;
  for (long long i = (long long)blockIdx.x * 256 + threadIdx.x; i < total;
       i += (long long)gridDim.x * 256)
    acc += emb[i];                       // stride % 64 == 0 -> fixed column per thread
  s[threadIdx.x] = acc;
  __syncthreads();
  if (threadIdx.x < 64) {
    float vsum = s[threadIdx.x] + s[threadIdx.x + 64] +
                 s[threadIdx.x + 128] + s[threadIdx.x + 192];
    atomicAdd(&out[threadIdx.x & 63], vsum);
  }
}

__global__ void zero_pool(float* p) {
  if (threadIdx.x < 192) p[threadIdx.x] = 0.0f;
}

// ---------------------------------------------------------------------------
// head: fused = relu(mean(pooled) @ fuse_W + fuse_b); cls = fused@cls_W+cls_b;
//       reg = sigmoid(fused@reg_W+reg_b). out[0..9]=cls, out[10]=reg.
// ---------------------------------------------------------------------------
__global__ __launch_bounds__(64) void head_kernel(
    const float* __restrict__ pooled, float invN,
    const float* __restrict__ fW, const float* __restrict__ fb,
    const float* __restrict__ cW, const float* __restrict__ cb,
    const float* __restrict__ rW, const float* __restrict__ rb,
    float* __restrict__ out) {
  __shared__ float fused[64];
  int j = threadIdx.x;
  float acc = fb[j];
  for (int i = 0; i < 192; ++i) acc += (pooled[i] * invN) * fW[i * 64 + j];
  fused[j] = fmaxf(acc, 0.0f);
  __syncthreads();
  if (j < 10) {
    float c = cb[j];
    for (int i = 0; i < 64; ++i) c += fused[i] * cW[i * 10 + j];
    out[j] = c;
  } else if (j == 10) {
    float r = rb[0];
    for (int i = 0; i < 64; ++i) r += fused[i] * rW[i];
    out[10] = 1.0f / (1.0f + __expf(-r));
  }
}

// ---------------------------------------------------------------------------
// Host orchestration
// ---------------------------------------------------------------------------
static inline int cdiv(long long a, int b) { return (int)((a + b - 1) / b); }

extern "C" void kernel_launch(void* const* d_in, const int* in_sizes, int n_in,
                              void* d_out, int out_size, void* d_ws, size_t ws_size,
                              hipStream_t stream) {
  (void)n_in; (void)out_size; (void)ws_size;
  const int N = in_sizes[0] / 1024;           // 50000
  const int dins[3]  = {1024, 256, 128};
  const int douts[3] = {256, 128, 64};
  const int lgDs[3]  = {8, 7, 6};

  const float* xs[3] = {(const float*)d_in[0], (const float*)d_in[1],
                        (const float*)d_in[2]};
  const int*   eis[3] = {(const int*)d_in[3], (const int*)d_in[4],
                         (const int*)d_in[5]};
  const int    Es[3] = {in_sizes[3] / 2, in_sizes[4] / 2, in_sizes[5] / 2};

  // per-layer parameter packs: base index 6 + 6*l  -> W, B, g, bt, m, v
  const float* Wl[3]; const float* Bl[3]; const float* gl[3];
  const float* btl[3]; const float* ml[3]; const float* vl[3];
  for (int l = 0; l < 3; ++l) {
    Wl[l]  = (const float*)d_in[6 + 6 * l + 0];
    Bl[l]  = (const float*)d_in[6 + 6 * l + 1];
    gl[l]  = (const float*)d_in[6 + 6 * l + 2];
    btl[l] = (const float*)d_in[6 + 6 * l + 3];
    ml[l]  = (const float*)d_in[6 + 6 * l + 4];
    vl[l]  = (const float*)d_in[6 + 6 * l + 5];
  }
  const float* fuse_W = (const float*)d_in[24];
  const float* fuse_b = (const float*)d_in[25];
  const float* cls_W  = (const float*)d_in[26];
  const float* cls_b  = (const float*)d_in[27];
  const float* reg_W  = (const float*)d_in[28];
  const float* reg_b  = (const float*)d_in[29];

  // workspace: two N x 256 fp32 ping-pong buffers + deg/dinv + pooled[192]
  float* ws0    = (float*)d_ws;
  float* ws1    = ws0 + (size_t)N * 256;
  float* deg    = ws1 + (size_t)N * 256;
  float* pooled = deg + N;

  zero_pool<<<1, 192, 0, stream>>>(pooled);

  for (int s = 0; s < 3; ++s) {
    const int E = Es[s];
    const int* src = eis[s];
    const int* dst = eis[s] + E;

    // dinv = rsqrt(deg + 1), shared by all 3 layers of this branch
    deg_init<<<cdiv(N, 256), 256, 0, stream>>>(deg, N);
    deg_accum<<<cdiv(E, 256), 256, 0, stream>>>(dst, E, deg);
    deg_rsqrt<<<cdiv(N, 256), 256, 0, stream>>>(deg, N);

    const float* X = xs[s];
    for (int l = 0; l < 3; ++l) {
      const int din = dins[l], dout = douts[l], lgD = lgDs[l];
      float* H   = (l & 1) ? ws1 : ws0;
      float* AGG = (l & 1) ? ws0 : ws1;
      float* Y   = H;   // H is dead once scatter finishes; BN reads AGG only

      dim3 ggrid(cdiv(N, MT), dout / NT);
      gcn_gemm_wmma_bf16<<<ggrid, 256, 0, stream>>>(
          X, Wl[l] + (size_t)s * din * dout, H, N, din, dout);

      long long nelem = (long long)N << lgD;
      agg_selfloop<<<cdiv(nelem, 256), 256, 0, stream>>>(H, deg, AGG, N, lgD);

      long long escat = (long long)E << lgD;
      agg_scatter<<<cdiv(escat, 256), 256, 0, stream>>>(H, deg, src, dst, AGG,
                                                        (long long)E, lgD);

      bias_bn_act<<<cdiv(nelem, 256), 256, 0, stream>>>(
          AGG, Bl[l] + (size_t)s * dout, gl[l] + (size_t)s * dout,
          btl[l] + (size_t)s * dout, ml[l] + (size_t)s * dout,
          vl[l] + (size_t)s * dout, Y, N, lgD, (l < 2) ? 1 : 0);
      X = Y;
    }
    // X now points at emb [N,64] (ws0); pool into pooled[s*64 ..]
    colsum64<<<256, 256, 0, stream>>>(X, pooled + s * 64, N);
  }

  head_kernel<<<1, 64, 0, stream>>>(pooled, 1.0f / (float)N,
                                    fuse_W, fuse_b, cls_W, cls_b,
                                    reg_W, reg_b, (float*)d_out);
}
